// Head_1443109011600
// MI455X (gfx1250) — matrix operand order repaired and checked by reading, not compiled, not run on hardware
//
#include <hip/hip_runtime.h>
#include <hip/hip_bf16.h>
#include <stdint.h>

// ---------------------------------------------------------------------------
// Causal single-head attention for MI455X (gfx1250), bf16 WMMA pipeline.
//   B=8, T=4096, C=1024, HS=64.  scale = C^-0.5 = 1/32.
//   Stage 2 uses block-cooperative, double-buffered K/V tiles in LDS filled by
//   GLOBAL_LOAD_ASYNC_TO_LDS_B128 (ASYNCcnt) when the builtin is available.
// ---------------------------------------------------------------------------

typedef __bf16 bf16;
typedef bf16  v16bf __attribute__((ext_vector_type(16)));
typedef bf16  v8bf  __attribute__((ext_vector_type(8)));
typedef float v8f   __attribute__((ext_vector_type(8)));
typedef unsigned int u32;
typedef int v4i_g __attribute__((vector_size(16)));   // builtin's V4i type

#define TT   4096
#define CC   1024
#define HSZ  64
#define NROW (8 * TT)            // 32768 flattened rows
#define QK_ELEMS (NROW * HSZ)    // 2097152 bf16 per q/k/v buffer

#define AS1 __attribute__((address_space(1)))
#define AS3 __attribute__((address_space(3)))

#if defined(__gfx1250__) && __has_builtin(__builtin_amdgcn_global_load_async_to_lds_b128)
#define HAVE_ASYNC 1
#else
#define HAVE_ASYNC 0
#endif

// 16-byte global -> LDS copy: async (ASYNCcnt) when available, else sync.
// AS1 pointer = virtual address; AS3 pointer = low 32 bits of the flat address
// (LDS aperture maps addr[31:0] directly to the LDS offset).
static __device__ __forceinline__ void cp16(void* lds, const void* g) {
#if HAVE_ASYNC
  __builtin_amdgcn_global_load_async_to_lds_b128(
      (AS1 v4i_g*)(uintptr_t)g, (AS3 v4i_g*)(u32)(uintptr_t)lds, 0, 0);
#else
  *(uint4*)lds = *(const uint4*)g;
#endif
}

static __device__ __forceinline__ v16bf join8(v8bf lo, v8bf hi) {
  return __builtin_shufflevector(lo, hi, 0,1,2,3,4,5,6,7,8,9,10,11,12,13,14,15);
}

// ---------------------------------------------------------------------------
// Kernel 0: fp32 weights -> bf16, pre-swizzled into WMMA B-fragment layout.
// wz[m][kk(32)][nt(4)][lane(32)][i(16)] =
//   W_m[(kk*32 + (i&7) + (lane>=16)*8 + (i>>3)*16) * 64 + nt*16 + (lane&15)]
// ---------------------------------------------------------------------------
__global__ void swz_weights(const float* __restrict__ Wk,
                            const float* __restrict__ Wq,
                            const float* __restrict__ Wv,
                            bf16* __restrict__ wz) {
  int idx = blockIdx.x * 256 + threadIdx.x;   // 0 .. 3*65536-1
  int i  = idx & 15;
  int L  = (idx >> 4) & 31;
  int nt = (idx >> 9) & 3;
  int kk = (idx >> 11) & 31;
  int m  = idx >> 16;
  const float* W = (m == 0) ? Wk : (m == 1) ? Wq : Wv;
  int ks = kk * 32 + (i & 7) + ((L >> 4) & 1) * 8 + (i >> 3) * 16;
  int ns = nt * 16 + (L & 15);
  wz[idx] = (bf16)W[ks * HSZ + ns];
}

// ---------------------------------------------------------------------------
// Kernel 1: q/k/v projections.  One wave per (16-row tile, matrix).
//   m=0 -> k (row-major), m=1 -> q (row-major), m=2 -> v (transposed [b][h][T])
// ---------------------------------------------------------------------------
__global__ void __launch_bounds__(256) qkv_proj(const float* __restrict__ x,
                                                const bf16* __restrict__ wz,
                                                bf16* __restrict__ qb,
                                                bf16* __restrict__ kb,
                                                bf16* __restrict__ vt) {
  const int lane = threadIdx.x & 31;
  const int wave = threadIdx.x >> 5;
  const int m    = blockIdx.y;
  const int tile = blockIdx.x * 8 + wave;     // 0..2047
  const int t0   = tile * 16;
  const int ln   = lane & 15;
  const int hi   = (lane >> 4) & 1;

  const bf16*  wm = wz + m * 65536;
  const float* xr = x + (size_t)(t0 + ln) * CC + hi * 8;

  v8f acc[4] = {};

  for (int kk = 0; kk < 32; ++kk) {
    const float* xp = xr + kk * 32;
    __builtin_prefetch(xp + 32, 0, 0);               // next x chunk -> caches
    // A fragment: 16x32 bf16 chunk of x
    v16bf a;
#pragma unroll
    for (int j = 0; j < 4; ++j) {
      float2 p0 = *(const float2*)(xp + 2 * j);        // K = hi*8 + 2j, 2j+1
      float2 p1 = *(const float2*)(xp + 16 + 2 * j);   // K = 16 + hi*8 + 2j, 2j+1
      a[2 * j + 0] = (bf16)p0.x;  a[2 * j + 1] = (bf16)p0.y;
      a[2 * j + 8] = (bf16)p1.x;  a[2 * j + 9] = (bf16)p1.y;
    }
    const bf16* wp = wm + (size_t)(kk * 2048) + lane * 16;
#pragma unroll
    for (int nt = 0; nt < 4; ++nt) {
      v16bf bfr = *(const v16bf*)(wp + nt * 512);
      acc[nt] = __builtin_amdgcn_wmma_f32_16x16x32_bf16(
          false, a, false, bfr, (short)0, acc[nt], false, false);
    }
  }

  // C/D layout: VGPR r, lane -> (row = r + hi*8, col = nt*16 + ln)
#pragma unroll
  for (int nt = 0; nt < 4; ++nt) {
#pragma unroll
    for (int r = 0; r < 8; ++r) {
      int row = t0 + r + hi * 8;
      int col = nt * 16 + ln;
      bf16 vvv = (bf16)acc[nt][r];
      if (m == 0)      kb[(size_t)row * HSZ + col] = vvv;
      else if (m == 1) qb[(size_t)row * HSZ + col] = vvv;
      else {
        int bb = row >> 12, tt = row & (TT - 1);
        vt[((size_t)(bb * HSZ + col) << 12) + tt] = vvv;
      }
    }
  }
}

// ---------------------------------------------------------------------------
// Kernel 2: flash attention.  Block = 8 waves = 8 consecutive query tiles
// (same batch).  All waves share one double-buffered K/V tile in LDS per
// 32-key step (8x global-traffic reduction + async copy/compute overlap).
// ---------------------------------------------------------------------------
__global__ void __launch_bounds__(256) attn(const bf16* __restrict__ qb,
                                            const bf16* __restrict__ kb,
                                            const bf16* __restrict__ vt,
                                            float* __restrict__ out) {
  __shared__ __align__(16) bf16 plds[8][16][48];    // per-wave P bounce, 12 KB
  __shared__ __align__(16) bf16 ktile[2][32][64];   // [buf][key][feat], 8 KB
  __shared__ __align__(16) bf16 vtile[2][64][32];   // [buf][head][key], 8 KB

  const int tid   = threadIdx.x;
  const int lane  = tid & 31;
  const int wave  = tid >> 5;
  const int ln    = lane & 15;
  const int hi    = (lane >> 4) & 1;
  const int qtile = blockIdx.x * 8 + wave;          // 0..2047
  const int b     = qtile >> 8;                     // 256 tiles per batch
  const int q0    = (qtile & 255) * 16;
  const size_t g0 = (size_t)b * TT + q0;

  const int smax      = q0 + 16;                                   // this wave
  const int blockSmax = (((blockIdx.x * 8 + 7) & 255) * 16) + 16;  // last wave

  const bf16* kbase = kb + ((size_t)b * TT) * HSZ;
  const bf16* vbase = vt + ((size_t)b << 18);       // b*HSZ*TT
  const char* kbB = (const char*)kbase;
  const char* vbB = (const char*)vbase;

  // Q A-fragments (16 rows x 64 features = two 16x32 fragments)
  const bf16* qp = qb + (g0 + ln) * HSZ;
  v16bf qa[2];
#pragma unroll
  for (int c = 0; c < 2; ++c) {
    v8bf lo = *(const v8bf*)(qp + c * 32 + hi * 16);
    v8bf h8 = *(const v8bf*)(qp + c * 32 + hi * 16 + 8);
    qa[c] = join8(lo, h8);
  }

  v8f o[4] = {};
  float mrow[8], lrow[8];
#pragma unroll
  for (int r = 0; r < 8; ++r) { mrow[r] = -1e30f; lrow[r] = 0.0f; }

  // cooperative K (32x64 bf16 = 4KB) + V (64x32 bf16 = 4KB) tile copy:
  // each of 256 threads moves one 16B chunk of each.
  auto issue_tile = [&](int bufi, int s0n) {
    cp16(&ktile[bufi][tid >> 3][(tid & 7) * 8],
         kbB + (size_t)(s0n + (tid >> 3)) * 128 + (tid & 7) * 16);
    cp16(&vtile[bufi][tid >> 2][(tid & 3) * 8],
         vbB + (size_t)(tid >> 2) * 8192 + (size_t)s0n * 2 + (tid & 3) * 16);
  };

  issue_tile(0, 0);

  for (int s0 = 0; s0 < blockSmax; s0 += 32) {
    const int  cur     = (s0 >> 5) & 1;
    const bool hasNext = (s0 + 32) < blockSmax;
    if (hasNext) issue_tile(cur ^ 1, s0 + 32);      // prefetch next tile
#if HAVE_ASYNC
    if (hasNext) asm volatile("s_wait_asynccnt 0x2" ::: "memory");
    else         asm volatile("s_wait_asynccnt 0x0" ::: "memory");
#endif
    __syncthreads();                                // current tile resident

    if (s0 < smax) {                                // wave-uniform causal skip
      // S = Q @ K^T for 2 key sub-tiles of 16 (B-frags from LDS)
      v8f st[2] = {};
#pragma unroll
      for (int t = 0; t < 2; ++t) {
        int ks = s0 + t * 16;
        if (ks < smax) {                            // wave-uniform
          const bf16* kp = &ktile[cur][t * 16 + ln][hi * 16];
#pragma unroll
          for (int c = 0; c < 2; ++c) {
            v16bf kf = *(const v16bf*)(kp + c * 32);
            st[t] = __builtin_amdgcn_wmma_f32_16x16x32_bf16(
                false, qa[c], false, kf, (short)0, st[t], false, false);
          }
        }
      }

      // scale + causal mask + online softmax (rows live in 16-lane halves)
#pragma unroll
      for (int r = 0; r < 8; ++r) {
        int rowq = q0 + r + hi * 8;
        float v0 = (s0 + ln      <= rowq) ? st[0][r] * 0.03125f : -1e30f;
        float v1 = (s0 + 16 + ln <= rowq) ? st[1][r] * 0.03125f : -1e30f;
        float mx = fmaxf(v0, v1);
#pragma unroll
        for (int off = 1; off < 16; off <<= 1)
          mx = fmaxf(mx, __shfl_xor(mx, off, 32));
        float mnew = fmaxf(mrow[r], mx);
        float corr = __expf(mrow[r] - mnew);
        float p0 = __expf(v0 - mnew);
        float p1 = __expf(v1 - mnew);
        float ps = p0 + p1;
#pragma unroll
        for (int off = 1; off < 16; off <<= 1)
          ps += __shfl_xor(ps, off, 32);
        lrow[r] = lrow[r] * corr + ps;
        mrow[r] = mnew;
#pragma unroll
        for (int ht = 0; ht < 4; ++ht) o[ht][r] *= corr;
        plds[wave][r + hi * 8][ln]      = (bf16)p0;
        plds[wave][r + hi * 8][16 + ln] = (bf16)p1;
      }

      // re-layout P (C/D layout) -> 16x32 A fragment through LDS
      asm volatile("s_wait_dscnt 0x0" ::: "memory");
      const bf16* pp  = &plds[wave][ln][hi * 16];
      v8bf  plo = *(const v8bf*)(pp);
      v8bf  phi = *(const v8bf*)(pp + 8);
      v16bf pa  = join8(plo, phi);

      // O += P @ V  (V tile transposed in LDS: contiguous B-fragment loads)
#pragma unroll
      for (int ht = 0; ht < 4; ++ht) {
        const bf16* vp = &vtile[cur][ht * 16 + ln][hi * 16];
        v16bf vf = *(const v16bf*)vp;
        o[ht] = __builtin_amdgcn_wmma_f32_16x16x32_bf16(
            false, pa, false, vf, (short)0, o[ht], false, false);
      }
    }

    __syncthreads();   // all reads of buf[cur] done before it is overwritten
  }

  // normalize and store fp32 output
#pragma unroll
  for (int ht = 0; ht < 4; ++ht) {
#pragma unroll
    for (int r = 0; r < 8; ++r) {
      size_t row = g0 + r + hi * 8;
      out[row * HSZ + ht * 16 + ln] = o[ht][r] / lrow[r];
    }
  }
}

// ---------------------------------------------------------------------------
extern "C" void kernel_launch(void* const* d_in, const int* in_sizes, int n_in,
                              void* d_out, int out_size, void* d_ws, size_t ws_size,
                              hipStream_t stream) {
  const float* x  = (const float*)d_in[0];
  const float* Wk = (const float*)d_in[1];
  const float* Wq = (const float*)d_in[2];
  const float* Wv = (const float*)d_in[3];
  float* out = (float*)d_out;

  // workspace layout (region order gives safe landing zones for masked-tail reads)
  bf16* qb = (bf16*)d_ws;            // 4 MB
  bf16* kb = qb + QK_ELEMS;          // 4 MB (tail reads spill into vt)
  bf16* vt = kb + QK_ELEMS;          // 4 MB (tail reads spill into wz)
  bf16* wz = vt + QK_ELEMS;          // 384 KB swizzled weights

  swz_weights<<<768, 256, 0, stream>>>(Wk, Wq, Wv, wz);
  qkv_proj<<<dim3(256, 3), 256, 0, stream>>>(x, wz, qb, kb, vt);
  attn<<<256, 256, 0, stream>>>(qb, kb, vt, out);
}
